// GraphConvBlock_15607911154153
// MI455X (gfx1250) — matrix operand order repaired and checked by reading, not compiled, not run on hardware
//
#include <hip/hip_runtime.h>
#include <hip/hip_bf16.h>

typedef _Float16 half16_t __attribute__((ext_vector_type(16)));
typedef _Float16 half8_t  __attribute__((ext_vector_type(8)));
typedef float    float8_t __attribute__((ext_vector_type(8)));
typedef float    float4_t __attribute__((ext_vector_type(4)));

#define NB   32
#define CCH  64
#define TT   512
#define VJ   25
#define PPOS (NB*TT*VJ)   /* 409600 */
#define NTB  (NB*TT)      /* 16384  */
#define NPR  (TT*VJ)      /* 12800 rows per n */
#define PADR 200          /* extra padded rows per n (100 front, 100 back) */
#define NPRP (NPR+PADR)   /* 13000 padded rows per n */
#define PADROWS (NB*NPRP) /* 416000 */

static __device__ __forceinline__ float8_t wmma16(half16_t a, half16_t b, float8_t c) {
  return __builtin_amdgcn_wmma_f32_16x16x32_f16(false, a, false, b, (short)0, c, false, false);
}
static __device__ __forceinline__ half16_t cat8(half8_t lo, half8_t hi) {
  return __builtin_shufflevector(lo, hi, 0,1,2,3,4,5,6,7,8,9,10,11,12,13,14,15);
}

// ---------------------------------------------------------------------------
// K0: BN1 + transpose x[N,C,T,V] -> inp_f16[P,64]; fused LayerNorm -> qn_f16[P,64]
// ---------------------------------------------------------------------------
__global__ __launch_bounds__(256) void k0_bn_ln(
    const float* __restrict__ x,
    const float* __restrict__ g1, const float* __restrict__ b1,
    const float* __restrict__ m1, const float* __restrict__ v1,
    const float* __restrict__ lg, const float* __restrict__ lb,
    _Float16* __restrict__ inp, _Float16* __restrict__ qn)
{
  __shared__ float s1s[64], h1s[64], lgs[64], lbs[64];
  int tid = threadIdx.x;
  if (tid < 64) {
    float iv = rsqrtf(v1[tid] + 1e-5f);
    float s  = g1[tid] * iv;
    s1s[tid] = s; h1s[tid] = b1[tid] - m1[tid] * s;
    lgs[tid] = lg[tid]; lbs[tid] = lb[tid];
  }
  __syncthreads();
  int p = blockIdx.x * 256 + tid;
  int n = p / NPR;
  int r = p - n * NPR;
  const float* xp = x + (size_t)n * (CCH * NPR) + r;
  float reg[64]; float sm = 0.f, sq = 0.f;
  #pragma unroll
  for (int c = 0; c < 64; c++) {
    float val = xp[(size_t)c * NPR];
    val = val * s1s[c] + h1s[c];
    reg[c] = val; sm += val; sq += val * val;
  }
  float mu  = sm * (1.f / 64.f);
  float var = sq * (1.f / 64.f) - mu * mu;
  float inv = rsqrtf(var + 1e-6f);
  _Float16* ip = inp + (size_t)p * 64;
  _Float16* qp = qn  + (size_t)p * 64;
  #pragma unroll
  for (int g = 0; g < 8; g++) {
    half8_t hv = {}, hq = {};
    #pragma unroll
    for (int j = 0; j < 8; j++) {
      float val = reg[g*8 + j];
      hv[j] = (_Float16)val;
      hq[j] = (_Float16)((val - mu) * inv * lgs[g*8 + j] + lbs[g*8 + j]);
    }
    *(half8_t*)(ip + g*8) = hv;
    *(half8_t*)(qp + g*8) = hq;
  }
}

// ---------------------------------------------------------------------------
// Templated WMMA GEMM: Out[P,NOUT](f16) = A[P,64](f16) @ B(+bias)
// B element (k=c, n=o) read at B[c*bsk + o*bsn]; block = 128 rows, 8 waves.
// NOUT compile-time -> immediate-offset stores; HB compile-time -> no branch.
// ---------------------------------------------------------------------------
template<int NOUT, bool HB>
__global__ __launch_bounds__(256) void gemm_t(
    const _Float16* __restrict__ A, const float* __restrict__ B,
    int bsk, int bsn, const float* __restrict__ bias,
    _Float16* __restrict__ Out)
{
  __shared__ _Float16 Bl[NOUT * 64];   // column-major [o][c], f16
  int tid = threadIdx.x;
  for (int idx = tid; idx < NOUT * 64; idx += 256) {
    int o = idx >> 6, c = idx & 63;
    Bl[o * 64 + c] = (_Float16)B[(size_t)c * bsk + (size_t)o * bsn];
  }
  __syncthreads();
  int wave = tid >> 5, lane = tid & 31;
  int p0 = blockIdx.x * 128 + wave * 16;
  const _Float16* arow = A + (size_t)(p0 + (lane & 15)) * 64;
  int koff = (lane & 16) ? 16 : 0;
  half16_t a0 = cat8(*(const half8_t*)(arow + koff),      *(const half8_t*)(arow + koff + 8));
  half16_t a1 = cat8(*(const half8_t*)(arow + 32 + koff), *(const half8_t*)(arow + 40 + koff));
  int rbase = p0 + ((lane & 16) ? 8 : 0);
  #pragma unroll
  for (int nt = 0; nt < (NOUT >> 4); nt++) {
    int ocol = nt * 16 + (lane & 15);
    const _Float16* bp = Bl + ocol * 64 + ((lane & 16) ? 16 : 0);
    half16_t b0 = *(const half16_t*)bp;
    half16_t b1 = *(const half16_t*)(bp + 32);
    float8_t acc = {0,0,0,0,0,0,0,0};
    acc = wmma16(a0, b0, acc);
    acc = wmma16(a1, b1, acc);
    float bb = HB ? bias[ocol] : 0.f;
    _Float16* op = Out + (size_t)rbase * NOUT + ocol;
    #pragma unroll
    for (int r2 = 0; r2 < 8; r2++)
      op[r2 * NOUT] = (_Float16)(acc[r2] + bb);
  }
}

// ---------------------------------------------------------------------------
// Combined K/V projection: one pass over A=inp, two weight sets in LDS.
// ---------------------------------------------------------------------------
__global__ __launch_bounds__(256) void gemm_kv(
    const _Float16* __restrict__ A,
    const float* __restrict__ Bk, const float* __restrict__ Bv,
    _Float16* __restrict__ OutK, _Float16* __restrict__ OutV)
{
  __shared__ _Float16 Bl[128 * 64];    // [o<64: wk | o>=64: wv][c]
  int tid = threadIdx.x;
  for (int idx = tid; idx < 128 * 64; idx += 256) {
    int o = idx >> 6, c = idx & 63;
    float w = (o < 64) ? Bk[c * 64 + o] : Bv[c * 64 + (o - 64)];
    Bl[o * 64 + c] = (_Float16)w;
  }
  __syncthreads();
  int wave = tid >> 5, lane = tid & 31;
  int p0 = blockIdx.x * 128 + wave * 16;
  const _Float16* arow = A + (size_t)(p0 + (lane & 15)) * 64;
  int koff = (lane & 16) ? 16 : 0;
  half16_t a0 = cat8(*(const half8_t*)(arow + koff),      *(const half8_t*)(arow + koff + 8));
  half16_t a1 = cat8(*(const half8_t*)(arow + 32 + koff), *(const half8_t*)(arow + 40 + koff));
  int rbase = p0 + ((lane & 16) ? 8 : 0);
  #pragma unroll
  for (int nt = 0; nt < 8; nt++) {
    int ocol = nt * 16 + (lane & 15);          // 0..127 across both outputs
    const _Float16* bp = Bl + ocol * 64 + ((lane & 16) ? 16 : 0);
    half16_t b0 = *(const half16_t*)bp;
    half16_t b1 = *(const half16_t*)(bp + 32);
    float8_t acc = {0,0,0,0,0,0,0,0};
    acc = wmma16(a0, b0, acc);
    acc = wmma16(a1, b1, acc);
    _Float16* op = (nt < 4)
        ? (OutK + (size_t)rbase * 64 + ocol)
        : (OutV + (size_t)rbase * 64 + (ocol - 64));
    #pragma unroll
    for (int r2 = 0; r2 < 8; r2++)
      op[r2 * 64] = (_Float16)acc[r2];
  }
}

// ---------------------------------------------------------------------------
// K2: graph aggregation  fagg[(n,t,w),co] = sum_{k,v} fc[(n,t,v),k*64+co]*A[k,v,w]
// ---------------------------------------------------------------------------
__global__ __launch_bounds__(256) void k2_agg(
    const _Float16* __restrict__ fc, const float* __restrict__ Amat,
    _Float16* __restrict__ fagg)
{
  __shared__ float fcl[25 * 192];
  __shared__ float Al[3 * 25 * 25];
  int tid = threadIdx.x;
  size_t b = blockIdx.x;
  const _Float16* src = fc + b * 25 * 192;
  for (int idx = tid; idx < 25 * 192; idx += 256) fcl[idx] = (float)src[idx];
  for (int idx = tid; idx < 1875;      idx += 256) Al[idx] = Amat[idx];
  __syncthreads();
  for (int idx = tid; idx < 64 * 25; idx += 256) {
    int co = idx & 63, w = idx >> 6;
    float acc = 0.f;
    #pragma unroll
    for (int k = 0; k < 3; k++) {
      const float* Ar = Al  + (k * 25) * 25 + w;
      const float* fr = fcl + k * 64 + co;
      #pragma unroll
      for (int v2 = 0; v2 < 25; v2++) acc += fr[v2 * 192] * Ar[v2 * 25];
    }
    fagg[(b * 25 + w) * 64 + co] = (_Float16)acc;
  }
}

// ---------------------------------------------------------------------------
// K3: attention core per (n,t): scores, softmax, att@V -> xh[P,64] f16
// ---------------------------------------------------------------------------
__global__ __launch_bounds__(256) void k3_attn(
    const _Float16* __restrict__ q, const _Float16* __restrict__ k,
    const _Float16* __restrict__ v, _Float16* __restrict__ xh)
{
  __shared__ float qL[25*64], kL[25*64], vL[25*64], aL[4*25*25];
  int tid = threadIdx.x;
  size_t base = (size_t)blockIdx.x * 25 * 64;
  for (int idx = tid; idx < 1600; idx += 256) {
    qL[idx] = (float)q[base + idx];
    kL[idx] = (float)k[base + idx];
    vL[idx] = (float)v[base + idx];
  }
  __syncthreads();
  for (int idx = tid; idx < 2500; idx += 256) {
    int h = idx / 625; int rem = idx - h * 625; int qi = rem / 25, kj = rem % 25;
    float acc = 0.f;
    #pragma unroll
    for (int d = 0; d < 16; d++) acc += qL[qi*64 + h*16 + d] * kL[kj*64 + h*16 + d];
    aL[idx] = acc * 0.25f;   // 1/sqrt(DK=16)
  }
  __syncthreads();
  if (tid < 100) {           // 4 heads * 25 query rows
    float* row = aL + tid * 25;
    float mx = row[0];
    #pragma unroll
    for (int j = 1; j < 25; j++) mx = fmaxf(mx, row[j]);
    float s = 0.f;
    #pragma unroll
    for (int j = 0; j < 25; j++) { float e = __expf(row[j] - mx); row[j] = e; s += e; }
    float is = 1.f / s;
    #pragma unroll
    for (int j = 0; j < 25; j++) row[j] *= is;
  }
  __syncthreads();
  for (int idx = tid; idx < 1600; idx += 256) {
    int qi = idx >> 6, col = idx & 63, h = col >> 4;
    const float* arow = aL + (h * 25 + qi) * 25;
    float acc = 0.f;
    #pragma unroll
    for (int kj = 0; kj < 25; kj++) acc += arow[kj] * vL[kj*64 + col];
    xh[base + idx] = (_Float16)acc;
  }
}

// ---------------------------------------------------------------------------
// K4: f = relu(BN2(((xh@fc_w + fa)*gate + fagg)*0.5)) -> f16, PADDED layout
// ---------------------------------------------------------------------------
__global__ __launch_bounds__(256) void k4_fc_fuse(
    const _Float16* __restrict__ xh, const float* __restrict__ fcw,
    const _Float16* __restrict__ inp, const _Float16* __restrict__ fagg,
    const float* __restrict__ gate_p,
    const float* __restrict__ g2, const float* __restrict__ b2,
    const float* __restrict__ m2, const float* __restrict__ v2,
    _Float16* __restrict__ fpad)
{
  __shared__ _Float16 Bl[64 * 64];
  int tid = threadIdx.x;
  for (int idx = tid; idx < 4096; idx += 256) {
    int o = idx >> 6, c = idx & 63;
    Bl[o * 64 + c] = (_Float16)fcw[c * 64 + o];
  }
  __syncthreads();
  float gate = gate_p[0];
  int wave = tid >> 5, lane = tid & 31;
  int p0 = blockIdx.x * 128 + wave * 16;
  int n  = (blockIdx.x * 128) / NPR;        // tiles never cross n (12800%128==0)
  int padoff = n * PADR + 100;
  const _Float16* arow = xh + (size_t)(p0 + (lane & 15)) * 64;
  int koff = (lane & 16) ? 16 : 0;
  half16_t a0 = cat8(*(const half8_t*)(arow + koff),      *(const half8_t*)(arow + koff + 8));
  half16_t a1 = cat8(*(const half8_t*)(arow + 32 + koff), *(const half8_t*)(arow + 40 + koff));
  int rbase = p0 + ((lane & 16) ? 8 : 0);
  #pragma unroll
  for (int nt = 0; nt < 4; nt++) {
    int ocol = nt * 16 + (lane & 15);
    const _Float16* bp = Bl + ocol * 64 + ((lane & 16) ? 16 : 0);
    float8_t acc = {0,0,0,0,0,0,0,0};
    acc = wmma16(a0, *(const half16_t*)bp,        acc);
    acc = wmma16(a1, *(const half16_t*)(bp + 32), acc);
    float s2 = g2[ocol] * rsqrtf(v2[ocol] + 1e-5f);
    float h2 = b2[ocol] - m2[ocol] * s2;
    const _Float16* ipp = inp  + (size_t)rbase * 64 + ocol;
    const _Float16* fgp = fagg + (size_t)rbase * 64 + ocol;
    _Float16*       op  = fpad + (size_t)(rbase + padoff) * 64 + ocol;
    #pragma unroll
    for (int r2 = 0; r2 < 8; r2++) {
      float val = acc[r2] + (float)ipp[r2 * 64];                 // + fa residual
      val = (val * gate + (float)fgp[r2 * 64]) * 0.5f;           // gated fusion
      val = fmaxf(val * s2 + h2, 0.f);                           // BN2 + ReLU
      op[r2 * 64] = (_Float16)val;
    }
  }
}

// ---------------------------------------------------------------------------
// zero the 100-row front/back pads of fpad (re-run every call: deterministic)
// ---------------------------------------------------------------------------
__global__ __launch_bounds__(256) void kpad_zero(_Float16* __restrict__ fpad)
{
  int g = blockIdx.x * 256 + threadIdx.x;   // one group = 8 halves
  if (g >= 6400 * 8) return;                // 6400 pad rows * 64 halves
  int r  = g >> 3;
  int c0 = (g & 7) * 8;
  int nblk = r / PADR; int loc = r - nblk * PADR;
  int prow = nblk * NPRP + (loc < 100 ? loc : NPR + loc);  // front : back pad
  half8_t z = {};
  *(half8_t*)(fpad + (size_t)prow * 64 + c0) = z;
}

// ---------------------------------------------------------------------------
// weight prep for tconv: W2t[co][dt*64+ci] = tconv_w[co,ci,dt]  (f16, [64][576])
// ---------------------------------------------------------------------------
__global__ __launch_bounds__(256) void kw_prep(const float* __restrict__ tw,
                                               _Float16* __restrict__ W2t)
{
  int idx = blockIdx.x * 256 + threadIdx.x;
  if (idx >= 64 * 576) return;
  int co = idx / 576; int dk = idx - co * 576;
  int dt = dk >> 6;   int ci = dk & 63;
  W2t[idx] = (_Float16)tw[((co * 64 + ci) * 9 + dt)];
}

// ---------------------------------------------------------------------------
// K5: temporal conv as GEMM 128x64x576 per position tile (padded-f shifted rows)
//     + bias + BN3 + x-residual + ReLU, LDS-transposed coalesced f32 output.
// ---------------------------------------------------------------------------
#define LROWS 328          /* 128 + 200 halo rows */
#define LSTR  72           /* LDS row stride in halves (16B aligned, no conflicts) */
__global__ __launch_bounds__(256) void k5_tconv(
    const _Float16* __restrict__ fpad, const _Float16* __restrict__ W2t,
    const float* __restrict__ x, const float* __restrict__ tb,
    const float* __restrict__ g3, const float* __restrict__ b3,
    const float* __restrict__ m3, const float* __restrict__ v3,
    float* __restrict__ out)
{
  __shared__ _Float16 fl[LROWS * LSTR];
  __shared__ float    twl[8 * 320];        // per-wave 16x16 f32 transpose, stride 20
  int tid = threadIdx.x;
  int p0 = blockIdx.x * 128;
  int n  = p0 / NPR;
  // contiguous padded source rows [p0 + n*PADR, +LROWS)
  const _Float16* src = fpad + (size_t)(p0 + n * PADR) * 64;
  for (int idx = tid; idx < LROWS * 8; idx += 256) {     // 8-half groups per row
    int row = idx >> 3; int c0 = (idx & 7) * 8;
    *(half8_t*)(fl + row * LSTR + c0) = *(const half8_t*)(src + (size_t)row * 64 + c0);
  }
  __syncthreads();
  int wave = tid >> 5, lane = tid & 31;
  int m0 = wave * 16;
  float8_t z8 = {0,0,0,0,0,0,0,0};
  float8_t acc[4] = {z8, z8, z8, z8};
  int koffA = (lane & 16) ? 16 : 0;
  int koffB = (lane & 16) ? 16 : 0;
  int arowbase = m0 + (lane & 15);
  for (int kc = 0; kc < 18; kc++) {        // K = 9*64 = 576 in chunks of 32
    int dt  = kc >> 1;
    int cib = ((kc & 1) << 5) + koffA;
    const _Float16* fr = fl + (arowbase + 25 * dt) * LSTR + cib;
    half16_t a = cat8(*(const half8_t*)fr, *(const half8_t*)(fr + 8));
    #pragma unroll
    for (int nt = 0; nt < 4; nt++) {
      int ocol = nt * 16 + (lane & 15);
      half16_t b = *(const half16_t*)(W2t + (size_t)ocol * 576 + kc * 32 + koffB);
      acc[nt] = wmma16(a, b, acc[nt]);
    }
  }
  float* myt = twl + wave * 320;
  int colw = lane & 15;
  int rb   = (lane & 16) ? 8 : 0;
  int jcol = lane >> 1;                    // 0..15 : output channel within tile
  int seg  = lane & 1;                     // 0/1   : 8-position segment
  int plocal = p0 - n * NPR + m0;          // (t*25+v) base of this wave's tile
  #pragma unroll
  for (int nt = 0; nt < 4; nt++) {
    {   // write phase: bias + BN3 applied per accumulator column
      int ocol = nt * 16 + colw;
      float s3 = g3[ocol] * rsqrtf(v3[ocol] + 1e-5f);
      float hh = b3[ocol] - m3[ocol] * s3 + tb[ocol] * s3;
      #pragma unroll
      for (int r2 = 0; r2 < 8; r2++)
        myt[colw * 20 + rb + r2] = acc[nt][r2] * s3 + hh;
    }
    __syncthreads();
    {   // read phase: coalesced residual + ReLU + store (full 64B lines per co)
      int co = nt * 16 + jcol;
      size_t xi = ((size_t)n * 64 + co) * NPR + (size_t)(plocal + seg * 8);
      const float* tsrc = myt + jcol * 20 + seg * 8;
      float4_t t0 = *(const float4_t*)(tsrc);
      float4_t t1 = *(const float4_t*)(tsrc + 4);
      float4_t x0 = *(const float4_t*)(x + xi);
      float4_t x1 = *(const float4_t*)(x + xi + 4);
      float4_t o0, o1;
      #pragma unroll
      for (int j = 0; j < 4; j++) {
        o0[j] = fmaxf(t0[j] + x0[j], 0.f);
        o1[j] = fmaxf(t1[j] + x1[j], 0.f);
      }
      *(float4_t*)(out + xi)     = o0;
      *(float4_t*)(out + xi + 4) = o1;
    }
    __syncthreads();
  }
}

// ---------------------------------------------------------------------------
extern "C" void kernel_launch(void* const* d_in, const int* in_sizes, int n_in,
                              void* d_out, int out_size, void* d_ws, size_t ws_size,
                              hipStream_t stream)
{
  (void)in_sizes; (void)n_in; (void)out_size; (void)ws_size;
  const float* x   = (const float*)d_in[0];
  const float* Am  = (const float*)d_in[1];
  const float* g1  = (const float*)d_in[2];
  const float* b1  = (const float*)d_in[3];
  const float* m1  = (const float*)d_in[4];
  const float* v1  = (const float*)d_in[5];
  const float* pw  = (const float*)d_in[6];
  const float* pb  = (const float*)d_in[7];
  const float* wq  = (const float*)d_in[8];
  const float* wk  = (const float*)d_in[9];
  const float* wv  = (const float*)d_in[10];
  const float* fcw = (const float*)d_in[11];
  const float* lg  = (const float*)d_in[12];
  const float* lb  = (const float*)d_in[13];
  const float* gate= (const float*)d_in[14];
  const float* g2  = (const float*)d_in[15];
  const float* b2  = (const float*)d_in[16];
  const float* m2  = (const float*)d_in[17];
  const float* v2  = (const float*)d_in[18];
  const float* tw  = (const float*)d_in[19];
  const float* tb  = (const float*)d_in[20];
  const float* g3  = (const float*)d_in[21];
  const float* b3  = (const float*)d_in[22];
  const float* m3  = (const float*)d_in[23];
  const float* v3  = (const float*)d_in[24];
  float* out = (float*)d_out;

  const size_t P = (size_t)PPOS;
  _Float16* ws16 = (_Float16*)d_ws;
  _Float16* inp  = ws16;                          // [P,64]  fa (BN1 out), f16
  _Float16* qn   = ws16 + P * 64;                 // [P,64]  LN(fa); reused as xh
  _Float16* big  = ws16 + P * 128;                // [P,192] f_c; then q|k|v
  _Float16* fagg = ws16 + P * 320;                // [P,64]  graph-agg output
  _Float16* fpad = ws16 + P * 384;                // [416000,64] padded fused f
  _Float16* W2t  = fpad + (size_t)PADROWS * 64;   // [64,576] tconv weights f16
  _Float16* qb = big;
  _Float16* kb = big + P * 64;
  _Float16* vb = big + P * 128;
  _Float16* xh = qn;                              // alias (qn dead after q-proj)

  kw_prep         <<<144,       256, 0, stream>>>(tw, W2t);
  kpad_zero       <<<200,       256, 0, stream>>>(fpad);
  k0_bn_ln        <<<PPOS/256,  256, 0, stream>>>(x, g1, b1, m1, v1, lg, lb, inp, qn);
  gemm_t<192,true><<<PPOS/128,  256, 0, stream>>>(inp, pw, 1, 64, pb, big);  // partconv
  k2_agg          <<<NTB,       256, 0, stream>>>(big, Am, fagg);            // graph agg
  gemm_t<64,false><<<PPOS/128,  256, 0, stream>>>(qn, wq, 64, 1, nullptr, qb);
  gemm_kv         <<<PPOS/128,  256, 0, stream>>>(inp, wk, wv, kb, vb);
  k3_attn         <<<NTB,       256, 0, stream>>>(qb, kb, vb, xh);
  k4_fc_fuse      <<<PPOS/128,  256, 0, stream>>>(xh, fcw, inp, fagg, gate,
                                                  g2, b2, m2, v2, fpad);
  k5_tconv        <<<PPOS/128,  256, 0, stream>>>(fpad, W2t, x, tb, g3, b3, m3, v3, out);
}
